// EncoderCell_56642028699793
// MI455X (gfx1250) — matrix operand order repaired and checked by reading, not compiled, not run on hardware
//
#include <hip/hip_runtime.h>

// ---------------------------------------------------------------------------
// Performer encoder cell for gfx1250 (MI455X)
// wave32, v_wmma_f32_16x16x32_bf16, async global->LDS staged GEMMs
// B=4, L=4096, D=1024, H=16, dh=64, m=256, scale_factor=4, Lq=1024
// ---------------------------------------------------------------------------

typedef __bf16 bf16;
typedef __attribute__((ext_vector_type(16))) __bf16 v16bf;
typedef __attribute__((ext_vector_type(8)))  float  v8f;

#define WB 4
#define WL 4096
#define WD 1024
#define WH 16
#define WDH 64
#define WM 256
#define WLQ 1024
#define SC_QK 0.35355339059327373f   /* dh^-0.25 = 64^-0.25 */

#define LDS_PAD 40                   /* row stride (elems) for 32-K LDS tiles */

union FragU { v16bf v; unsigned u[8]; bf16 h[16]; };

static __device__ inline v8f zero_v8f() {
    v8f z;
#pragma unroll
    for (int i = 0; i < 8; ++i) z[i] = 0.0f;
    return z;
}

// A fragment: 16(M) x 32(K), bf16 row-major, lda elements.
// lanes 0-15 hold K {0..15}, lanes 16-31 hold K {16..31}.
static __device__ inline v16bf load_a_bf16(const bf16* __restrict__ A, int lda) {
    const int lane = threadIdx.x & 31;
    const int half = lane >> 4;
    const int r    = lane & 15;
    const bf16* row = A + (long long)r * lda + half * 16;
    FragU f;
#pragma unroll
    for (int i = 0; i < 4; ++i) f.u[i]     = *(const unsigned*)(row + 2 * i);
#pragma unroll
    for (int i = 0; i < 4; ++i) f.u[4 + i] = *(const unsigned*)(row + 8 + 2 * i);
    return f.v;
}

// B fragment: 32(K) x 16(N) from "packed" storage [N, K] (K contiguous).
// VGPR i holds K = 2i + 16*half for column n = lane&15.
static __device__ inline v16bf load_b_packed(const bf16* __restrict__ Bp, int ldk) {
    const int lane = threadIdx.x & 31;
    const int half = lane >> 4;
    const int n    = lane & 15;
    const bf16* col = Bp + (long long)n * ldk + half * 16;
    FragU f;
#pragma unroll
    for (int i = 0; i < 8; ++i) f.u[i] = *(const unsigned*)(col + 2 * i);
    return f.v;
}

static __device__ inline v8f wmma_bf16(v16bf a, v16bf b, v8f c) {
    return __builtin_amdgcn_wmma_f32_16x16x32_bf16(false, a, false, b,
                                                   (short)0, c, false, false);
}

// Async global -> LDS copy of 16 bytes (per lane), tracked by ASYNCcnt.
static __device__ inline void async_b128(unsigned lds_off, unsigned long long gaddr) {
    asm volatile("global_load_async_to_lds_b128 %0, %1, off"
                 :: "v"(lds_off), "v"(gaddr) : "memory");
}

// ---------------------------------------------------------------------------
// Prep kernels
// ---------------------------------------------------------------------------

// x f32 -> xb bf16 (bulk convert)
__global__ __launch_bounds__(256) void xb_kernel(const float* __restrict__ x,
                                                 bf16* __restrict__ xb) {
    const long long idx = (long long)blockIdx.x * 256 + threadIdx.x;
    xb[idx] = (bf16)x[idx];
}

// avg_pool1d(x, 4) -> pooled bf16 [B*Lq, D]
__global__ __launch_bounds__(256) void pool_kernel(const float* __restrict__ x,
                                                   bf16* __restrict__ pooledb) {
    const long long idx = (long long)blockIdx.x * 256 + threadIdx.x; // B*Lq*D
    const long long rowD = idx >> 10;        // b*Lq + lq
    const int dcol = (int)(idx & 1023);
    const long long b  = rowD >> 10;         // Lq = 1024
    const long long lq = rowD & 1023;
    const float* p = x + ((b * WL + lq * 4) << 10) + dcol;
    const float v = 0.25f * (p[0] + p[1024] + p[2048] + p[3072]);
    pooledb[idx] = (bf16)v;
}

// W [K=1024, N=1024] f32 row-major -> WT bf16 [N, K] (packed-B layout)
__global__ __launch_bounds__(256) void wt_kernel(const float* __restrict__ W,
                                                 bf16* __restrict__ WT) {
    const int idx = blockIdx.x * 256 + threadIdx.x; // 1M
    const int n = idx >> 10, k = idx & 1023;
    WT[idx] = (bf16)W[k * WD + n];
}

// proj [m, dh] f32 -> bf16 scaled by sc (already packed-B: [N=m, K=dh])
__global__ __launch_bounds__(256) void proj_kernel(const float* __restrict__ proj,
                                                   bf16* __restrict__ projb) {
    const int idx = blockIdx.x * 256 + threadIdx.x; // 16384
    projb[idx] = (bf16)(proj[idx] * SC_QK);
}

// ---------------------------------------------------------------------------
// LDS-tiled WMMA GEMM: C = A[M,K]bf16 @ Bp[N,K]bf16^T + bias.
// Block tile 128x128, K staged 32/iter, double-buffered async global->LDS.
// 8 waves: wave tile 32(M) x 64(N) = 2x4 WMMA accumulators.
// MODE 0: f32 row-major; MODE 1: bf16 row-major;
// MODE 2: bf16 per-head transposed vT[((b*16+h)*64+d)*Lseq + l]
// ---------------------------------------------------------------------------
template <int MODE>
__global__ __launch_bounds__(256) void gemm_kernel(const bf16* __restrict__ Ab,
                                                   const bf16* __restrict__ Bp,
                                                   const float* __restrict__ bias,
                                                   void* __restrict__ Cout,
                                                   int M, int N, int K, int Lseq) {
    __shared__ bf16 sA[2][128 * LDS_PAD];
    __shared__ bf16 sB[2][128 * LDS_PAD];

    const int wave = threadIdx.x >> 5;
    const int lane = threadIdx.x & 31;
    const int nblk = N >> 7;
    const int row0 = (blockIdx.x / nblk) << 7;
    const int col0 = (blockIdx.x % nblk) << 7;

    const int wr = wave >> 1;   // 0..3 : 32-row group
    const int wc = wave & 1;    // 0..1 : 64-col group

    const unsigned baseA = (unsigned)(size_t)(&sA[0][0]);
    const unsigned baseB = (unsigned)(size_t)(&sB[0][0]);
    const unsigned bufB  = (unsigned)(128 * LDS_PAD * 2); // bytes per buffer

    // Stage one 128x32 tile of A and of B into LDS buffer p (4 async/wave-lane).
    auto stage = [&](int kk, int p) {
        const unsigned t = threadIdx.x;
#pragma unroll
        for (int j = 0; j < 2; ++j) {
            const unsigned c   = t + 256u * j;       // chunk 0..511
            const unsigned row = c >> 2;             // 0..127
            const unsigned ko  = (c & 3) * 8;        // K offset (elems)
            const unsigned lo  = row * (LDS_PAD * 2) + ko * 2; // bytes
            async_b128(baseA + p * bufB + lo,
                       (unsigned long long)(Ab + (size_t)(row0 + row) * K + kk + ko));
            async_b128(baseB + p * bufB + lo,
                       (unsigned long long)(Bp + (size_t)(col0 + row) * K + kk + ko));
        }
    };

    v8f acc[2][4];
#pragma unroll
    for (int mt = 0; mt < 2; ++mt)
#pragma unroll
        for (int nt = 0; nt < 4; ++nt) acc[mt][nt] = zero_v8f();

    const int NS = K >> 5;
    stage(0, 0);
    for (int s = 0; s < NS; ++s) {
        const int p = s & 1;
        __syncthreads();  // all waves done reading buffer 1-p
        const int kn = (s + 1 < NS) ? ((s + 1) << 5) : 0; // last: dummy restage
        stage(kn, 1 - p);
        asm volatile("s_wait_asynccnt 0x4" ::: "memory"); // stage s complete (this wave)
        __syncthreads();  // stage s complete (all waves)

        v16bf af[2];
#pragma unroll
        for (int mt = 0; mt < 2; ++mt)
            af[mt] = load_a_bf16(&sA[p][(wr * 32 + mt * 16) * LDS_PAD], LDS_PAD);
#pragma unroll
        for (int nt = 0; nt < 4; ++nt) {
            const v16bf b = load_b_packed(&sB[p][(wc * 64 + nt * 16) * LDS_PAD], LDS_PAD);
            acc[0][nt] = wmma_bf16(af[0], b, acc[0][nt]);
            acc[1][nt] = wmma_bf16(af[1], b, acc[1][nt]);
        }
    }
    asm volatile("s_wait_asynccnt 0x0" ::: "memory");

    const int cn = lane & 15;
    const int mb = (lane >> 4) * 8;
#pragma unroll
    for (int mt = 0; mt < 2; ++mt) {
#pragma unroll
        for (int nt = 0; nt < 4; ++nt) {
            const int col = col0 + wc * 64 + nt * 16 + cn;
            const float bv = bias[col];
#pragma unroll
            for (int e = 0; e < 8; ++e) {
                const int row = row0 + wr * 32 + mt * 16 + mb + e;
                const float val = acc[mt][nt][e] + bv;
                if (MODE == 0) {
                    ((float*)Cout)[(long long)row * N + col] = val;
                } else if (MODE == 1) {
                    ((bf16*)Cout)[(long long)row * N + col] = (bf16)val;
                } else {
                    const int b = row / Lseq, l = row % Lseq;
                    const int h = col >> 6, d = col & 63;
                    ((bf16*)Cout)[((long long)((b * WH + h) * WDH + d)) * Lseq + l] =
                        (bf16)val;
                }
            }
        }
    }
}

// ---------------------------------------------------------------------------
// phi: rows are token-head pairs (b,l,h) with A = T[row*64 .. +64] bf16.
// u = A @ projb^T (scale folded into projb); phi = exp(u - rowmax)/16 + eps.
// (The reference's -0.5|t|^2 term is a per-row constant and cancels against
//  the per-row max subtraction.)
// TRANS=1 -> fkT [b,h,m,L]; TRANS=0 -> fq [b,h,Lq,m].
// ---------------------------------------------------------------------------
template <bool TRANS>
__global__ __launch_bounds__(256) void phi_kernel(const bf16* __restrict__ T,
                                                  const bf16* __restrict__ projb,
                                                  bf16* __restrict__ out, int Lseq) {
    __shared__ float us[16 * 256];
    __shared__ float pmax[256];
    __shared__ float rmax[16];

    const int wave = threadIdx.x >> 5;
    const int lane = threadIdx.x & 31;
    const long long row0 = (long long)blockIdx.x * 16;
    const bf16* A = T + row0 * WDH;

    const v16bf a0 = load_a_bf16(A, WDH);
    const v16bf a1 = load_a_bf16(A + 32, WDH);

    const int cn = lane & 15;
    const int mb = (lane >> 4) * 8;
#pragma unroll
    for (int j = 0; j < 2; ++j) {
        const int ct = wave * 2 + j;  // column tile 0..15
        const v16bf b0 = load_b_packed(projb + ct * 16 * WDH, WDH);
        const v16bf b1 = load_b_packed(projb + ct * 16 * WDH + 32, WDH);
        v8f c = zero_v8f();
        c = wmma_bf16(a0, b0, c);
        c = wmma_bf16(a1, b1, c);
#pragma unroll
        for (int e = 0; e < 8; ++e)
            us[(mb + e) * 256 + ct * 16 + cn] = c[e];
    }
    __syncthreads();

    {   // per-row max, two-stage
        const int t = threadIdx.x, r = t >> 4, seg = t & 15;
        float mx = -1e30f;
#pragma unroll
        for (int i = 0; i < 16; ++i) mx = fmaxf(mx, us[r * 256 + seg * 16 + i]);
        pmax[t] = mx;
    }
    __syncthreads();
    if (threadIdx.x < 16) {
        const int r = threadIdx.x;
        float mx = -1e30f;
#pragma unroll
        for (int i = 0; i < 16; ++i) mx = fmaxf(mx, pmax[r * 16 + i]);
        rmax[r] = mx;
    }
    __syncthreads();

    {   // exp + store
        const int t = threadIdx.x, r = t >> 4, seg = t & 15;
        const long long grow = row0 + r;
        const int h = (int)(grow & 15);
        const long long bl = grow >> 4;
        const int l = (int)(bl % Lseq);
        const int b = (int)(bl / Lseq);
        const float mx = rmax[r];
#pragma unroll
        for (int i = 0; i < 16; ++i) {
            const int col = seg * 16 + i;
            const float v = __expf(us[r * 256 + col] - mx) * 0.0625f + 1e-6f;
            long long idx;
            if (TRANS) idx = ((long long)((b * WH + h) * WM + col)) * Lseq + l;
            else       idx = ((long long)((b * WH + h) * Lseq + l)) * WM + col;
            out[idx] = (bf16)v;
        }
    }
}

// ksum[b,h,m] = sum_l fkT[b,h,m,l]; one wave per row.
__global__ __launch_bounds__(256) void ksum_kernel(const bf16* __restrict__ fkT,
                                                   float* __restrict__ ksum) {
    const int wave = threadIdx.x >> 5, lane = threadIdx.x & 31;
    const long long row = (long long)blockIdx.x * 8 + wave; // B*H*m = 16384
    const bf16* p = fkT + row * WL;
    float s = 0.0f;
    for (int i = lane; i < WL; i += 32) s += (float)p[i];
#pragma unroll
    for (int off = 16; off > 0; off >>= 1) s += __shfl_xor(s, off, 32);
    if (lane == 0) ksum[row] = s;
}

// kv[b,h] = fkT[b,h] [256,L] @ v[b,h] [L,64]; wave tile 32(m) x 64(dh).
// Store kvT bf16 [b,h,64,256] (packed-B for the num GEMM).
__global__ __launch_bounds__(256) void kv_kernel(const bf16* __restrict__ fkT,
                                                 const bf16* __restrict__ vT,
                                                 bf16* __restrict__ kvT) {
    const int wave = threadIdx.x >> 5, lane = threadIdx.x & 31;
    const int gid = blockIdx.x * 8 + wave;    // 512 wave-tiles
    const int slice = gid >> 3;               // b*H + h (8 waves per slice)
    const int mg = gid & 7;                   // 32-row m group

    const bf16* A  = fkT + ((long long)slice * WM + mg * 32) * WL;
    const bf16* Bp = vT  + (long long)slice * WDH * WL;

    v8f acc[2][4];
#pragma unroll
    for (int mt = 0; mt < 2; ++mt)
#pragma unroll
        for (int nt = 0; nt < 4; ++nt) acc[mt][nt] = zero_v8f();

    for (int kk = 0; kk < WL; kk += 32) {
        v16bf af[2];
#pragma unroll
        for (int mt = 0; mt < 2; ++mt)
            af[mt] = load_a_bf16(A + (long long)(mt * 16) * WL + kk, WL);
#pragma unroll
        for (int nt = 0; nt < 4; ++nt) {
            const v16bf b = load_b_packed(Bp + (long long)(nt * 16) * WL + kk, WL);
            acc[0][nt] = wmma_bf16(af[0], b, acc[0][nt]);
            acc[1][nt] = wmma_bf16(af[1], b, acc[1][nt]);
        }
    }

    const int cn = lane & 15, mb = (lane >> 4) * 8;
#pragma unroll
    for (int mt = 0; mt < 2; ++mt) {
#pragma unroll
        for (int nt = 0; nt < 4; ++nt) {
            const int d = nt * 16 + cn;
#pragma unroll
            for (int e = 0; e < 8; ++e) {
                const int mrow = mg * 32 + mt * 16 + mb + e;
                kvT[((long long)slice * WDH + d) * WM + mrow] = (bf16)acc[mt][nt][e];
            }
        }
    }
}

// num = fq[b,h][Lq,256] @ kv; den = fq . ksum; attn = num/den -> bf16 [B*Lq, D]
__global__ __launch_bounds__(256) void attn_kernel(const bf16* __restrict__ fqb,
                                                   const bf16* __restrict__ kvT,
                                                   const float* __restrict__ ksum,
                                                   bf16* __restrict__ attnb) {
    __shared__ float pden[256];
    __shared__ float rden[32];

    const int wave = threadIdx.x >> 5, lane = threadIdx.x & 31;
    const int blocksPerSlice = WLQ >> 5;                // 32
    const int slice = blockIdx.x / blocksPerSlice;      // b*H + h
    const int rb    = blockIdx.x % blocksPerSlice;
    const int row0  = rb << 5;                          // 32 Lq rows / block

    const bf16* A  = fqb + (long long)slice * WLQ * WM;
    const bf16* Bp = kvT + (long long)slice * WDH * WM;
    const float* ks = ksum + slice * WM;

    const int mtl = wave >> 2;  // 0..1
    const int nt  = wave & 3;   // 0..3
    const bf16* Aw = A + (long long)(row0 + mtl * 16) * WM;
    const bf16* Bw = Bp + nt * 16 * WM;

    v8f acc = zero_v8f();
#pragma unroll
    for (int kk = 0; kk < WM; kk += 32) {
        const v16bf a = load_a_bf16(Aw + kk, WM);
        const v16bf b = load_b_packed(Bw + kk, WM);
        acc = wmma_bf16(a, b, acc);
    }

    {   // den partials: 8 threads per row, 32 elems each
        const int t = threadIdx.x, r = t >> 3, seg = t & 7;
        const bf16* fr = A + (long long)(row0 + r) * WM + seg * 32;
        float s = 0.0f;
#pragma unroll
        for (int i = 0; i < 32; ++i) s += (float)fr[i] * ks[seg * 32 + i];
        pden[t] = s;
    }
    __syncthreads();
    if (threadIdx.x < 32) {
        const int r = threadIdx.x;
        float s = 0.0f;
#pragma unroll
        for (int i = 0; i < 8; ++i) s += pden[r * 8 + i];
        rden[r] = 1.0f / s;
    }
    __syncthreads();

    const int cn = lane & 15, mb = (lane >> 4) * 8;
    const int h = slice & 15, b = slice >> 4;
    const int d = nt * 16 + cn;
#pragma unroll
    for (int e = 0; e < 8; ++e) {
        const int rl = mtl * 16 + mb + e;  // local row 0..31
        const int l = row0 + rl;
        const float v = acc[e] * rden[rl];
        attnb[((long long)(b * WLQ + l)) * WD + h * WDH + d] = (bf16)v;
    }
}

// exact GELU + LayerNorm, one block per row of y [4096, 1024]
__global__ __launch_bounds__(256) void gelu_ln_kernel(const float* __restrict__ y,
                                                      const float* __restrict__ gamma,
                                                      const float* __restrict__ beta,
                                                      float* __restrict__ out) {
    __shared__ float red[256];
    const long long row = blockIdx.x;
    const float* yr = y + row * WD;

    float g[4];
    float s = 0.0f, s2 = 0.0f;
#pragma unroll
    for (int i = 0; i < 4; ++i) {
        const int c = threadIdx.x + i * 256;
        const float xv = yr[c];
        const float v = 0.5f * xv * (1.0f + erff(xv * 0.70710678118654752f));
        g[i] = v; s += v; s2 += v * v;
    }

    red[threadIdx.x] = s;
    __syncthreads();
    for (int off = 128; off > 0; off >>= 1) {
        if (threadIdx.x < off) red[threadIdx.x] += red[threadIdx.x + off];
        __syncthreads();
    }
    const float mean = red[0] * (1.0f / 1024.0f);
    __syncthreads();

    red[threadIdx.x] = s2;
    __syncthreads();
    for (int off = 128; off > 0; off >>= 1) {
        if (threadIdx.x < off) red[threadIdx.x] += red[threadIdx.x + off];
        __syncthreads();
    }
    const float m2 = red[0] * (1.0f / 1024.0f);
    const float var = m2 - mean * mean;
    const float rstd = rsqrtf(var + 1e-5f);

#pragma unroll
    for (int i = 0; i < 4; ++i) {
        const int c = threadIdx.x + i * 256;
        out[row * WD + c] = (g[i] - mean) * rstd * gamma[c] + beta[c];
    }
}

// ---------------------------------------------------------------------------
// Launch
// ---------------------------------------------------------------------------
extern "C" void kernel_launch(void* const* d_in, const int* in_sizes, int n_in,
                              void* d_out, int out_size, void* d_ws, size_t ws_size,
                              hipStream_t stream) {
    (void)in_sizes; (void)n_in; (void)out_size; (void)ws_size;

    const float* x     = (const float*)d_in[0];
    const float* Wq    = (const float*)d_in[1];
    const float* bq    = (const float*)d_in[2];
    const float* Wk    = (const float*)d_in[3];
    const float* bk    = (const float*)d_in[4];
    const float* Wv    = (const float*)d_in[5];
    const float* bv    = (const float*)d_in[6];
    const float* Wo    = (const float*)d_in[7];
    const float* bo    = (const float*)d_in[8];
    const float* proj  = (const float*)d_in[9];
    const float* gamma = (const float*)d_in[10];
    const float* beta  = (const float*)d_in[11];
    float* out = (float*)d_out;

    char* ws = (char*)d_ws;
    size_t off = 0;
    auto alloc = [&](size_t bytes) -> char* {
        char* p = ws + off;
        off = (off + bytes + 255) & ~(size_t)255;
        return p;
    };

    bf16* xb      = (bf16*)alloc((size_t)WB * WL * WD * 2);           // 32 MB
    bf16* pooledb = (bf16*)alloc((size_t)WB * WLQ * WD * 2);          // 8 MB
    bf16* WqT     = (bf16*)alloc((size_t)WD * WD * 2);
    bf16* WkT     = (bf16*)alloc((size_t)WD * WD * 2);
    bf16* WvT     = (bf16*)alloc((size_t)WD * WD * 2);
    bf16* WoT     = (bf16*)alloc((size_t)WD * WD * 2);
    bf16* projb   = (bf16*)alloc((size_t)WM * WDH * 2);
    bf16* qb      = (bf16*)alloc((size_t)WB * WLQ * WD * 2);          // 8 MB
    bf16* kb      = (bf16*)alloc((size_t)WB * WL * WD * 2);           // 32 MB
    bf16* vT      = (bf16*)alloc((size_t)WB * WH * WDH * WL * 2);     // 32 MB
    bf16* fqb     = (bf16*)alloc((size_t)WB * WH * WLQ * WM * 2);     // 32 MB
    bf16* fkT     = (bf16*)alloc((size_t)WB * WH * WM * WL * 2);      // 128 MB
    float* ksum   = (float*)alloc((size_t)WB * WH * WM * 4);
    bf16* kvT     = (bf16*)alloc((size_t)WB * WH * WDH * WM * 2);     // 2 MB
    bf16* attnb   = (bf16*)alloc((size_t)WB * WLQ * WD * 2);          // 8 MB
    float* yf     = (float*)alloc((size_t)WB * WLQ * WD * 4);         // 16 MB

    // --- prep ---
    xb_kernel<<<(WB * WL * WD) / 256, 256, 0, stream>>>(x, xb);
    pool_kernel<<<(WB * WLQ * WD) / 256, 256, 0, stream>>>(x, pooledb);
    wt_kernel<<<(WD * WD) / 256, 256, 0, stream>>>(Wq, WqT);
    wt_kernel<<<(WD * WD) / 256, 256, 0, stream>>>(Wk, WkT);
    wt_kernel<<<(WD * WD) / 256, 256, 0, stream>>>(Wv, WvT);
    wt_kernel<<<(WD * WD) / 256, 256, 0, stream>>>(Wo, WoT);
    proj_kernel<<<(WM * WDH) / 256, 256, 0, stream>>>(proj, projb);

    // --- projections: 128x128 block tiles ---
    const int Mq = WB * WLQ;   // 4096
    const int Mk = WB * WL;    // 16384
    gemm_kernel<1><<<(Mq / 128) * (WD / 128), 256, 0, stream>>>(
        pooledb, WqT, bq, qb, Mq, WD, WD, WLQ);
    gemm_kernel<1><<<(Mk / 128) * (WD / 128), 256, 0, stream>>>(
        xb, WkT, bk, kb, Mk, WD, WD, WL);
    gemm_kernel<2><<<(Mk / 128) * (WD / 128), 256, 0, stream>>>(
        xb, WvT, bv, vT, Mk, WD, WD, WL);

    // --- feature maps ---
    phi_kernel<false><<<(WB * WLQ * WH) / 16, 256, 0, stream>>>(qb, projb, fqb, WLQ);
    phi_kernel<true><<<(WB * WL * WH) / 16, 256, 0, stream>>>(kb, projb, fkT, WL);

    // --- linear attention contractions ---
    ksum_kernel<<<(WB * WH * WM) / 8, 256, 0, stream>>>(fkT, ksum);
    kv_kernel<<<(WB * WH * 8) / 8, 256, 0, stream>>>(fkT, vT, kvT);
    attn_kernel<<<(WB * WH) * (WLQ / 32), 256, 0, stream>>>(fqb, kvT, ksum, attnb);

    // --- output projection + GELU + LayerNorm ---
    gemm_kernel<0><<<(Mq / 128) * (WD / 128), 256, 0, stream>>>(
        attnb, WoT, bo, yf, Mq, WD, WD, 0);
    gelu_ln_kernel<<<WB * WLQ, 256, 0, stream>>>(yf, gamma, beta, out);
}
